// ParallelLlamaRingAttentionRmPad_68899865362685
// MI455X (gfx1250) — matrix operand order repaired and checked by reading, not compiled, not run on hardware
//
#include <hip/hip_runtime.h>
#include <math.h>

// ---------------------------------------------------------------------------
// Types
// ---------------------------------------------------------------------------
typedef __attribute__((ext_vector_type(16))) __bf16 v16bf;
typedef __attribute__((ext_vector_type(8)))  float  v8f;
typedef unsigned short bfraw;

struct alignas(16) U4 { unsigned x, y, z, w; };
struct alignas(16) F4 { float x, y, z, w; };
struct alignas(8)  S4 { bfraw x, y, z, w; };
union  F16B { v16bf v; U4 u[2]; };
union  U4S  { U4 u; unsigned short s[8]; };

#define HIDc   4096
#define HQc    32
#define HKVc   8
#define Dc     128
#define QKVNc  6144   // (HQ + 2*HKV) * D

__device__ __forceinline__ bfraw f2b(float f) {
  union { float f; unsigned u; } x; x.f = f;
  unsigned r = x.u + 0x7FFFu + ((x.u >> 16) & 1u);   // round-to-nearest-even
  return (bfraw)(r >> 16);
}
__device__ __forceinline__ float b2f(bfraw b) {
  union { unsigned u; float f; } x; x.u = ((unsigned)b) << 16;
  return x.f;
}

// ---------------------------------------------------------------------------
// Fragment loaders (wave32 CDNA5 WMMA layouts, 16-bit data)
// A 16x32: lane half h: V0-3 = K[h*8 .. h*8+7], V4-7 = K[16+h*8 .. +7]
// B 32x16 (stored N-major, contiguous K): lane half h: V0-3 = K[h*8 .. h*8+7], V4-7 = K[16+h*8 .. +7]
// `row` must already include (lane & 15). kElem is the K offset in elements.
// ---------------------------------------------------------------------------
__device__ __forceinline__ v16bf load_fragA(const char* base, int stride, int row,
                                            int kElem, int lane) {
  const char* p = base + row * stride + kElem * 2 + ((lane >> 4) << 4);
  F16B f;
  f.u[0] = *(const U4*)(p);
  f.u[1] = *(const U4*)(p + 32);
  return f.v;
}
__device__ __forceinline__ v16bf load_fragB(const char* base, int stride, int row,
                                            int kElem, int lane) {
  const char* p = base + row * stride + kElem * 2 + ((lane >> 4) << 4);
  F16B f;
  f.u[0] = *(const U4*)(p);
  f.u[1] = *(const U4*)(p + 32);
  return f.v;
}

// ---------------------------------------------------------------------------
// f32 -> bf16 conversion (vectorized x4)
// ---------------------------------------------------------------------------
__global__ void f32_to_bf16_vec4(const float* __restrict__ in,
                                 bfraw* __restrict__ out, long n4) {
  long i = (long)blockIdx.x * blockDim.x + threadIdx.x;
  if (i >= n4) return;
  F4 v = ((const F4*)in)[i];
  S4 o = { f2b(v.x), f2b(v.y), f2b(v.z), f2b(v.w) };
  ((S4*)out)[i] = o;
}

// ---------------------------------------------------------------------------
// f32 [K][N] -> bf16 [N][K] transpose (32x32 LDS tiles)
// ---------------------------------------------------------------------------
__global__ __launch_bounds__(256) void transpose_f32_bf16(
    const float* __restrict__ in, bfraw* __restrict__ out, int K, int N) {
  __shared__ float tile[32][33];
  int n0 = blockIdx.x * 32;
  int k0 = blockIdx.y * 32;
  int tx = threadIdx.x, ty = threadIdx.y;
  for (int i = ty; i < 32; i += 8)
    tile[i][tx] = in[(size_t)(k0 + i) * N + n0 + tx];
  __syncthreads();
  for (int i = ty; i < 32; i += 8)
    out[(size_t)(n0 + i) * K + k0 + tx] = f2b(tile[tx][i]);
}

// ---------------------------------------------------------------------------
// bf16 GEMM: C[M][N] = A[M][K] (row-major) * Bt[N][K] (N-major)
// block = 256 threads (8 waves), tile 128x128.
// K staged 64-wide, double-buffered LDS, global loads prefetched into
// registers before the barrier so they overlap the current chunk's WMMAs.
// One __syncthreads per 64-K chunk.
// ---------------------------------------------------------------------------
template <bool OUT_BF16>
__global__ __launch_bounds__(256) void gemm_bf16_kernel(
    const bfraw* __restrict__ A, const bfraw* __restrict__ Bt,
    void* __restrict__ Cout, int M, int N, int K) {
  (void)M;
  extern __shared__ char smem[];
  const int BUFB = 128 * 144;                 // 18432 B per matrix per stage

  const int tid  = threadIdx.x;
  const int lane = tid & 31;
  const int wid  = tid >> 5;
  const int waveM = wid & 3;      // 4 sub-tiles of 32 rows
  const int waveN = wid >> 2;     // 2 sub-tiles of 64 cols
  const int rlo  = lane & 15;
  const int hgr  = lane >> 4;
  const size_t m0 = (size_t)blockIdx.y * 128;
  const size_t n0 = (size_t)blockIdx.x * 128;

  v8f zero = {0.f, 0.f, 0.f, 0.f, 0.f, 0.f, 0.f, 0.f};
  v8f acc[2][4];
  for (int i = 0; i < 2; ++i)
    for (int j = 0; j < 4; ++j) acc[i][j] = zero;

  // each thread moves 4 x 16B for A and for B per 64-K chunk
  // slot = tid + it*256 -> row = slot>>3 (0..127), seg = slot&7 (16B units)
  U4 pa[4], pb[4];

  auto gload = [&](int kc) {
#pragma unroll
    for (int it = 0; it < 4; ++it) {
      int slot = tid + it * 256;
      int row = slot >> 3, seg = slot & 7;
      pa[it] = *(const U4*)(A  + (m0 + row) * (size_t)K + kc + seg * 8);
      pb[it] = *(const U4*)(Bt + (n0 + row) * (size_t)K + kc + seg * 8);
    }
  };
  auto lstore = [&](int buf) {
    char* la = smem + (buf ? 2 * BUFB : 0);
    char* lb = la + BUFB;
#pragma unroll
    for (int it = 0; it < 4; ++it) {
      int slot = tid + it * 256;
      int row = slot >> 3, seg = slot & 7;
      *(U4*)(la + row * 144 + seg * 16) = pa[it];
      *(U4*)(lb + row * 144 + seg * 16) = pb[it];
    }
  };

  const int nChunks = K >> 6;
  gload(0);
  lstore(0);

  for (int c = 0; c < nChunks; ++c) {
    const int cur = c & 1;
    char* la = smem + (cur ? 2 * BUFB : 0);
    char* lb = la + BUFB;
    if (c + 1 < nChunks) gload((c + 1) << 6);   // overlap with WMMAs below
    __syncthreads();                             // stage `cur` ready
#pragma unroll
    for (int kk = 0; kk < 64; kk += 32) {
      v16bf af[2], bfm[4];
#pragma unroll
      for (int i = 0; i < 2; ++i)
        af[i] = load_fragA(la, 144, waveM * 32 + i * 16 + rlo, kk, lane);
#pragma unroll
      for (int j = 0; j < 4; ++j)
        bfm[j] = load_fragB(lb, 144, waveN * 64 + j * 16 + rlo, kk, lane);
#pragma unroll
      for (int i = 0; i < 2; ++i)
#pragma unroll
        for (int j = 0; j < 4; ++j)
          acc[i][j] = __builtin_amdgcn_wmma_f32_16x16x32_bf16(
              false, af[i], false, bfm[j], (short)0, acc[i][j], false, false);
    }
    if (c + 1 < nChunks) lstore(cur ^ 1);
  }

  // epilogue: C tile layout -> lane holds col = rlo, rows r + 8*hgr
  for (int i = 0; i < 2; ++i)
    for (int j = 0; j < 4; ++j)
      for (int r = 0; r < 8; ++r) {
        size_t row = m0 + waveM * 32 + i * 16 + r + 8 * hgr;
        size_t col = n0 + waveN * 64 + j * 16 + rlo;
        float v = acc[i][j][r];
        if (OUT_BF16) ((bfraw*)Cout)[row * (size_t)N + col] = f2b(v);
        else          ((float*)Cout)[row * (size_t)N + col] = v;
      }
}

// ---------------------------------------------------------------------------
// RoPE in-place on bf16 qkv buffer [T][6144]; heads 0..31 = Q, 32..39 = K
// ---------------------------------------------------------------------------
__global__ void rope_kernel(bfraw* __restrict__ qkv,
                            const int* __restrict__ pos_ids, int T) {
  long idx = (long)blockIdx.x * blockDim.x + threadIdx.x;
  long total = (long)T * 40 * 64;
  if (idx >= total) return;
  int  i  = (int)(idx & 63);
  int  hh = (int)((idx >> 6) % 40);
  long t  = idx / (64 * 40);
  int col = (hh < HQc) ? hh * Dc : HQc * Dc + (hh - HQc) * Dc;
  bfraw* p = qkv + t * QKVNc + col;
  float posv = (float)pos_ids[t];
  float freq = powf(10000.0f, -(float)i * (1.0f / 64.0f));
  float ang  = posv * freq;
  float s = sinf(ang), c = cosf(ang);
  float x1 = b2f(p[i]), x2 = b2f(p[i + 64]);
  p[i]      = f2b(x1 * c - x2 * s);
  p[i + 64] = f2b(x2 * c + x1 * s);
}

// ---------------------------------------------------------------------------
// Causal GQA flash attention. 1 block = (batch b, head h, 64-row q block).
// 4 waves; wave w owns q rows [w*16, w*16+16). D = 128.
// ---------------------------------------------------------------------------
__global__ __launch_bounds__(128) void attn_kernel(
    const bfraw* __restrict__ qkv, bfraw* __restrict__ O, int L, int nQB) {
  extern __shared__ char smem[];
  char* Qs  = smem;                 // 64 rows x 272 B  (128 bf16 + pad)
  char* Ks  = Qs  + 64 * 272;       // 64 rows x 272 B
  char* Vts = Ks  + 64 * 272;       // 128 rows (d) x 144 B (64 bf16 k + pad)
  char* Ps  = Vts + 128 * 144;      // 4 waves x 16 rows x 144 B

  const int tid  = threadIdx.x;
  const int lane = tid & 31;
  const int wid  = tid >> 5;
  const int rlo  = lane & 15;
  const int hgr  = lane >> 4;

  int bid = blockIdx.x;
  int qb  = bid % nQB;
  int h   = (bid / nQB) % HQc;
  int b   = bid / (nQB * HQc);
  const int hk = h >> 2;            // GQA group of 4

  // stage Q tile (64 x 128 bf16)
  {
    int rw = tid >> 4, sg = tid & 15;
    for (int it = 0; it < 8; ++it) {
      int row = rw + it * 8;
      size_t tok = (size_t)b * L + (size_t)qb * 64 + row;
      U4 q = *(const U4*)(qkv + tok * QKVNc + h * Dc + sg * 8);
      *(U4*)(Qs + row * 272 + sg * 16) = q;
    }
  }

  float mrow[8], lrow[8];
  v8f zero = {0.f, 0.f, 0.f, 0.f, 0.f, 0.f, 0.f, 0.f};
  v8f acc[8];
  for (int r = 0; r < 8; ++r) { mrow[r] = -INFINITY; lrow[r] = 0.f; }
  for (int dt = 0; dt < 8; ++dt) acc[dt] = zero;

  const float scale = 0.08838834764831845f;   // 1/sqrt(128)

  for (int kb = 0; kb <= qb; ++kb) {
    __syncthreads();   // previous iteration's Vts/Ps reads done
    // stage K (row-major) and V (transposed d-major) tiles
    {
      int rw = tid >> 4, sg = tid & 15;
      for (int it = 0; it < 8; ++it) {
        int row = rw + it * 8;
        size_t tok = (size_t)b * L + (size_t)kb * 64 + row;
        U4 kv = *(const U4*)(qkv + tok * QKVNc + HQc * Dc + hk * Dc + sg * 8);
        *(U4*)(Ks + row * 272 + sg * 16) = kv;
        U4S vv;
        vv.u = *(const U4*)(qkv + tok * QKVNc + (HQc + HKVc) * Dc + hk * Dc + sg * 8);
        for (int e = 0; e < 8; ++e) {
          int d = sg * 8 + e;
          *(bfraw*)(Vts + d * 144 + row * 2) = vv.s[e];
        }
      }
    }
    __syncthreads();

    // S = Q K^T  (16 q rows x 64 k cols per wave)
    v8f s[4];
    for (int n = 0; n < 4; ++n) s[n] = zero;
    for (int kc = 0; kc < 128; kc += 32) {
      v16bf aq = load_fragA(Qs, 272, wid * 16 + rlo, kc, lane);
      for (int n = 0; n < 4; ++n) {
        v16bf bk = load_fragB(Ks, 272, n * 16 + rlo, kc, lane);
        s[n] = __builtin_amdgcn_wmma_f32_16x16x32_bf16(
            false, aq, false, bk, (short)0, s[n], false, false);
      }
    }

    // online softmax (per row r + 8*hgr; cols spread over 16 lanes)
    char* Pw = Ps + wid * (16 * 144);
    for (int r = 0; r < 8; ++r) {
      int qloc = wid * 16 + r + 8 * hgr;
      float vals[4];
      for (int n = 0; n < 4; ++n) {
        float v = (float)s[n][r] * scale;
        if (kb == qb) {
          int kloc = n * 16 + rlo;
          if (kloc > qloc) v = -INFINITY;
        }
        vals[n] = v;
      }
      float rmax = fmaxf(fmaxf(vals[0], vals[1]), fmaxf(vals[2], vals[3]));
      for (int off = 1; off < 16; off <<= 1)
        rmax = fmaxf(rmax, __shfl_xor(rmax, off, 32));
      float newm = fmaxf(mrow[r], rmax);
      float fac  = __expf(mrow[r] - newm);
      mrow[r] = newm;
      float psum = 0.f;
      int prow = r + 8 * hgr;
      for (int n = 0; n < 4; ++n) {
        float pv = __expf(vals[n] - newm);
        psum += pv;
        *(bfraw*)(Pw + prow * 144 + (n * 16 + rlo) * 2) = f2b(pv);
      }
      for (int off = 1; off < 16; off <<= 1)
        psum += __shfl_xor(psum, off, 32);
      lrow[r] = lrow[r] * fac + psum;
      for (int dt = 0; dt < 8; ++dt) acc[dt][r] *= fac;
    }
    __syncthreads();   // P visible (cross-lane), V reads ordered

    // O += P * V   (P: 16x64, V^T in LDS: 128 x 64)
    for (int pkc = 0; pkc < 64; pkc += 32) {
      v16bf ap = load_fragA(Pw, 144, rlo, pkc, lane);
      for (int dt = 0; dt < 8; ++dt) {
        v16bf bv = load_fragB(Vts, 144, dt * 16 + rlo, pkc, lane);
        acc[dt] = __builtin_amdgcn_wmma_f32_16x16x32_bf16(
            false, ap, false, bv, (short)0, acc[dt], false, false);
      }
    }
  }

  // normalize and write O (bf16, [T][4096])
  for (int r = 0; r < 8; ++r) {
    float inv = 1.0f / lrow[r];
    size_t tok = (size_t)b * L + (size_t)qb * 64 + wid * 16 + r + 8 * hgr;
    for (int dt = 0; dt < 8; ++dt) {
      int d = dt * 16 + rlo;
      O[tok * (size_t)(HQc * Dc) + h * Dc + d] = f2b((float)acc[dt][r] * inv);
    }
  }
}

// ---------------------------------------------------------------------------
// Host launcher
// ---------------------------------------------------------------------------
extern "C" void kernel_launch(void* const* d_in, const int* in_sizes, int n_in,
                              void* d_out, int out_size, void* d_ws, size_t ws_size,
                              hipStream_t stream) {
  (void)n_in; (void)out_size; (void)ws_size;
  const float* hs   = (const float*)d_in[0];
  const int*   pos  = (const int*)d_in[1];
  const int*   cu   = (const int*)d_in[2]; (void)cu;
  const float* wqkv = (const float*)d_in[3];
  const float* wo   = (const float*)d_in[4];

  const int T  = in_sizes[0] / HIDc;     // 4096
  const int nB = in_sizes[2] - 1;        // 4
  const int L  = T / nB;                 // 1024

  char* ws = (char*)d_ws;
  size_t off = 0;
  bfraw* Xb    = (bfraw*)(ws + off); off += (size_t)T * HIDc * 2;         // 32 MB
  bfraw* WqkvT = (bfraw*)(ws + off); off += (size_t)QKVNc * HIDc * 2;     // 48 MB
  bfraw* WoT   = (bfraw*)(ws + off); off += (size_t)HIDc * HIDc * 2;      // 32 MB
  bfraw* QKVb  = (bfraw*)(ws + off); off += (size_t)T * QKVNc * 2;        // 48 MB
  bfraw* Ob    = (bfraw*)(ws + off); off += (size_t)T * HIDc * 2;         // 32 MB

  // 1) convert hidden_states to bf16
  long n4 = (long)T * HIDc / 4;
  f32_to_bf16_vec4<<<(unsigned)((n4 + 255) / 256), 256, 0, stream>>>(hs, Xb, n4);

  // 2) transpose weights to N-major bf16
  dim3 tb(32, 8);
  transpose_f32_bf16<<<dim3(QKVNc / 32, HIDc / 32), tb, 0, stream>>>(wqkv, WqkvT, HIDc, QKVNc);
  transpose_f32_bf16<<<dim3(HIDc / 32, HIDc / 32),  tb, 0, stream>>>(wo,   WoT,   HIDc, HIDc);

  // 3) QKV = X @ Wqkv  (bf16 out)  -- double-buffered LDS, 72 KB
  size_t gemm_lds = 4 * 128 * 144;
  gemm_bf16_kernel<true><<<dim3(QKVNc / 128, T / 128), 256, gemm_lds, stream>>>(
      Xb, WqkvT, (void*)QKVb, T, QKVNc, HIDc);

  // 4) RoPE on q,k in place
  long nr = (long)T * 40 * 64;
  rope_kernel<<<(unsigned)((nr + 255) / 256), 256, 0, stream>>>(QKVb, pos, T);

  // 5) causal GQA flash attention -> Ob
  int nQB = L / 64;
  size_t attn_lds = 64 * 272 + 64 * 272 + 128 * 144 + 4 * 16 * 144;  // 62464 B
  attn_kernel<<<nB * HQc * nQB, 128, attn_lds, stream>>>(QKVb, Ob, L, nQB);

  // 6) out = O @ Wo  (f32 out)
  gemm_bf16_kernel<false><<<dim3(HIDc / 128, T / 128), 256, gemm_lds, stream>>>(
      Ob, WoT, d_out, T, HIDc, HIDc);
}
